// PES_15238543966920
// MI455X (gfx1250) — matrix operand order repaired and checked by reading, not compiled, not run on hardware
//
#include <hip/hip_runtime.h>
#include <math.h>

// ---------------- CDNA5 WMMA types / helpers ----------------
typedef __attribute__((ext_vector_type(16))) _Float16 v16h;
typedef __attribute__((ext_vector_type(8)))  float    v8f;

#define NWV   7
#define NANGC 13
#define NORB  84
#define NLH   128
#define SROW  96            // padded row stride for S / dS (16B aligned, 91 used)
#define XLD   104           // LDS ld (halves) for 16x96 activation tile
#define HLD   136           // LDS ld (halves) for 16x128 activation tile
#define W1TLD 104           // LDS ld (halves) for W1^T (128 rows x 96 cols)
#define W2TLD 136           // LDS ld (halves) for W2^T (128 rows x 128 cols)
#define WHLD  136           // LDS ld (halves) for row-major W copies
#define ZLD   132           // LDS ld (floats) for 16x128 f32 tiles
#define CUTR  4.5f
#define PI_F  3.14159265358979323846f

static __device__ __forceinline__ float sigm(float x) { return 1.f / (1.f + __expf(-x)); }

static __device__ __forceinline__ v8f wmma32(v16h a, v16h b, v8f c) {
  // D = A(16x32 f16) * B(32x16 f16) + C(16x16 f32)
  return __builtin_amdgcn_wmma_f32_16x16x32_f16(false, a, false, b, (short)0, c, false, false);
}

// A operand from LDS (row-major 16 x K f16, row stride ldm halves).
// Layout (ISA 7.12.2, 16-bit A 16x32): lanes 0-15 row=lane, elems 0..7 -> K=k0..k0+7,
// elems 8..15 -> K=k0+16..k0+23; lanes 16-31 same rows, K offset +8.
static __device__ __forceinline__ v16h ldsA(const _Float16* buf, int ldm, int k0, int lane) {
  v16h a;
  int m = lane & 15, hi = (lane >> 4) & 1;
  const _Float16* row = buf + m * ldm;
  int ka = k0 + hi * 8, kb = k0 + 16 + hi * 8;
#pragma unroll
  for (int e = 0; e < 8; ++e) { a[e] = row[ka + e]; a[8 + e] = row[kb + e]; }
  return a;
}

// B operand from LDS where the fragment's K runs CONTIGUOUSLY per row:
// Wl[col][k] (leading dim ld halves). b[e] = Wl[col*ld + k0+8h+e], b[8+e] = Wl[col*ld + k0+16+8h+e].
// two runs of 8 contiguous halves -> 2x ds_load_b128, rows bank-staggered by the padded ld.
static __device__ __forceinline__ v16h lBc(const _Float16* Wl, int ld, int k0, int colbase, int lane) {
  v16h b;
  int j = colbase + (lane & 15);
  int kb = k0 + ((lane >> 4) & 1) * 8;
  const _Float16* row = Wl + j * ld + kb;
#pragma unroll
  for (int e = 0; e < 8; ++e) { b[e] = row[e]; b[8 + e] = row[16 + e]; }
  return b;
}

// Store 16x16 f32 accumulator tile to LDS row-major buffer (+optional bias[globalCol]).
// Layout (ISA 7.12.2): lane col = lane&15, rows r + 8*(lane>=16).
static __device__ __forceinline__ void stD(float* buf, int ldm, int colbase, int lane, v8f acc, const float* bias) {
  int n = colbase + (lane & 15);
  int mb = ((lane >> 4) & 1) * 8;
  float bb = bias ? bias[n] : 0.f;
#pragma unroll
  for (int r = 0; r < 8; ++r) buf[(mb + r) * ldm + n] = acc[r] + bb;
}

template <int KCH>
static __device__ __forceinline__ v8f gemm16(const _Float16* A, int lda, const _Float16* Wl, int ld,
                                             int colbase, int lane) {
  v8f acc = {};
#pragma unroll
  for (int kk = 0; kk < KCH; ++kk)
    acc = wmma32(ldsA(A, lda, kk * 32, lane), lBc(Wl, ld, kk * 32, colbase, lane), acc);
  return acc;
}

// ---------------- init / zero ----------------
__global__ void __launch_bounds__(256) k_init(float* __restrict__ sz /* S0,S1 contiguous */,
                                              float* __restrict__ dc1, float* __restrict__ outp,
                                              float* __restrict__ c0a, const float* __restrict__ coeff,
                                              const int* __restrict__ ls, int n) {
  long t = (long)blockIdx.x * blockDim.x + threadIdx.x;
  long zn = 2L * n * SROW;
  if (t < zn) sz[t] = 0.f;
  if (t < 7L * n) dc1[t] = 0.f;
  if (t < 4L * n + 1) outp[t] = 0.f;   // E, forces(3N), per-atom E(N)
  if (t < 7L * n) {
    int i = (int)(t / 7), k = (int)(t - 7L * i);
    c0a[t] = coeff[ls[i] * NWV + k];
  }
}

// ---------------- per-pair geometry ----------------
static __device__ __forceinline__ void pair_geom(const float* cart, int i, int j, int sj,
                                                 const float* rs, const float* inta,
                                                 float r[3], float& d, float& fc,
                                                 float rad[NWV], float u[NANGC]) {
  r[0] = cart[i * 3 + 0] - cart[j * 3 + 0];
  r[1] = cart[i * 3 + 1] - cart[j * 3 + 1];
  r[2] = cart[i * 3 + 2] - cart[j * 3 + 2];
  d = sqrtf(r[0] * r[0] + r[1] * r[1] + r[2] * r[2]);
  float g = 0.5f * cosf(d * (PI_F / CUTR)) + 0.5f;
  fc = g * g;
#pragma unroll
  for (int k = 0; k < NWV; ++k) {
    float t = d - rs[sj * NWV + k];
    rad[k] = __expf(inta[sj * NWV + k] * t * t);
  }
  u[0] = 1.f; u[1] = r[0]; u[2] = r[1]; u[3] = r[2];
#pragma unroll
  for (int a = 0; a < 3; ++a)
#pragma unroll
    for (int b = 0; b < 3; ++b) u[4 + 3 * a + b] = r[a] * r[b];
}

static __device__ __forceinline__ void load_row96(const float* __restrict__ src, float* dst) {
  const float4* s4 = reinterpret_cast<const float4*>(src);
#pragma unroll
  for (int q = 0; q < 24; ++q) {
    float4 v = s4[q];
    dst[q * 4 + 0] = v.x; dst[q * 4 + 1] = v.y; dst[q * 4 + 2] = v.z; dst[q * 4 + 3] = v.w;
  }
}

// S[i,m,k] += ang[m]*rad[k]*catom[j,k]   (ang = fc*u)
__global__ void __launch_bounds__(256) k_pair_accum(const float* __restrict__ cart, const int* __restrict__ ai,
                                                    const int* __restrict__ nspec, const float* __restrict__ rs,
                                                    const float* __restrict__ inta, const float* __restrict__ catom,
                                                    float* __restrict__ S, int npair) {
  int p = blockIdx.x * 256 + threadIdx.x;
  if (p >= npair) return;
  int i = ai[p], j = ai[npair + p], sj = nspec[p];
  float r[3], d, fc, rad[NWV], u[NANGC];
  pair_geom(cart, i, j, sj, rs, inta, r, d, fc, rad, u);
  float* Si = S + (size_t)i * SROW;
#pragma unroll
  for (int k = 0; k < NWV; ++k) {
    float w = catom[(size_t)j * NWV + k] * rad[k] * fc;
#pragma unroll
    for (int m = 0; m < NANGC; ++m) unsafeAtomicAdd(&Si[m * NWV + k], u[m] * w);
  }
}

// dc1[j,k] += sum_m dS1[i,m,k] * orbital[m,k]
__global__ void __launch_bounds__(256) k_pair_bwdA(const float* __restrict__ cart, const int* __restrict__ ai,
                                                   const int* __restrict__ nspec, const float* __restrict__ rs,
                                                   const float* __restrict__ inta, const float* __restrict__ dS1,
                                                   float* __restrict__ dc1, int npair) {
  int p = blockIdx.x * 256 + threadIdx.x;
  if (p >= npair) return;
  int i = ai[p], j = ai[npair + p], sj = nspec[p];
  float r[3], d, fc, rad[NWV], u[NANGC];
  pair_geom(cart, i, j, sj, rs, inta, r, d, fc, rad, u);
  float row[SROW];
  load_row96(dS1 + (size_t)i * SROW, row);
#pragma unroll
  for (int k = 0; k < NWV; ++k) {
    float acc = 0.f;
#pragma unroll
    for (int m = 0; m < NANGC; ++m) acc += row[m * NWV + k] * u[m];
    unsafeAtomicAdd(&dc1[(size_t)j * NWV + k], acc * fc * rad[k]);
  }
}

// forces: dorb = dS1*c1 + dS0*c0 -> chain to rij, d; F = -grad
__global__ void __launch_bounds__(256) k_pair_bwdB(const float* __restrict__ cart, const int* __restrict__ ai,
                                                   const int* __restrict__ nspec, const float* __restrict__ rs,
                                                   const float* __restrict__ inta, const float* __restrict__ dS1,
                                                   const float* __restrict__ dS0, const float* __restrict__ c1,
                                                   const float* __restrict__ c0a, float* __restrict__ F, int npair) {
  int p = blockIdx.x * 256 + threadIdx.x;
  if (p >= npair) return;
  int i = ai[p], j = ai[npair + p], sj = nspec[p];
  float r[3], d, fc, rad[NWV], u[NANGC];
  pair_geom(cart, i, j, sj, rs, inta, r, d, fc, rad, u);
  float rowA[SROW], rowB[SROW];
  load_row96(dS1 + (size_t)i * SROW, rowA);
  load_row96(dS0 + (size_t)i * SROW, rowB);
  float c1k[NWV], c0k[NWV];
#pragma unroll
  for (int k = 0; k < NWV; ++k) { c1k[k] = c1[(size_t)j * NWV + k]; c0k[k] = c0a[(size_t)j * NWV + k]; }
  float dang[NANGC], drad[NWV];
#pragma unroll
  for (int k = 0; k < NWV; ++k) drad[k] = 0.f;
#pragma unroll
  for (int m = 0; m < NANGC; ++m) {
    float da = 0.f;
#pragma unroll
    for (int k = 0; k < NWV; ++k) {
      float val = rowA[m * NWV + k] * c1k[k] + rowB[m * NWV + k] * c0k[k];
      da += val * rad[k];
      drad[k] += val * fc * u[m];
    }
    dang[m] = da;
  }
  float dfc = 0.f;
#pragma unroll
  for (int m = 0; m < NANGC; ++m) dfc += dang[m] * u[m];
  float dr[3];
#pragma unroll
  for (int c = 0; c < 3; ++c) {
    float s = dang[1 + c];
#pragma unroll
    for (int b = 0; b < 3; ++b) s += dang[4 + 3 * c + b] * r[b] + dang[4 + 3 * b + c] * r[b];
    dr[c] = fc * s;
  }
  float g = 0.5f * cosf(d * (PI_F / CUTR)) + 0.5f;
  float dd = dfc * (-g * (PI_F / CUTR) * sinf(d * (PI_F / CUTR)));
#pragma unroll
  for (int k = 0; k < NWV; ++k)
    dd += drad[k] * rad[k] * 2.f * inta[sj * NWV + k] * (d - rs[sj * NWV + k]);
  float invd = (d > 1e-12f) ? (1.f / d) : 0.f;
#pragma unroll
  for (int c = 0; c < 3; ++c) {
    float gr = dr[c] + dd * r[c] * invd;  // dE/dcart_i contribution
    unsafeAtomicAdd(&F[(size_t)i * 3 + c], -gr);
    unsafeAtomicAdd(&F[(size_t)j * 3 + c], gr);
  }
}

// ---------------- contraction: density[i,o] = sum_m (sum_k S[i,m,k]*Wc[L(m),k,o])^2 ----------------
__global__ void __launch_bounds__(128) k_contract(const float* __restrict__ S, const float* __restrict__ Wc,
                                                  float* __restrict__ den, int natom) {
  __shared__ float sWc[3 * NWV * NORB];
  __shared__ float sS[NANGC * NWV];
  int i = blockIdx.x, tid = threadIdx.x;
  for (int idx = tid; idx < 3 * NWV * NORB; idx += 128) sWc[idx] = Wc[idx];
  for (int idx = tid; idx < NANGC * NWV; idx += 128) sS[idx] = S[(size_t)i * SROW + idx];
  __syncthreads();
  if (tid < NORB) {
    float acc = 0.f;
#pragma unroll
    for (int m = 0; m < NANGC; ++m) {
      int L = (m == 0) ? 0 : ((m < 4) ? 1 : 2);
      float h = 0.f;
#pragma unroll
      for (int k = 0; k < NWV; ++k) h += sS[m * NWV + k] * sWc[L * (NWV * NORB) + k * NORB + tid];
      acc += h * h;
    }
    den[(size_t)i * NORB + tid] = acc;
  }
}

// dS[i,m,k] = sum_o 2*H[i,m,o]*ddens[i,o]*Wc[L(m),k,o]
__global__ void __launch_bounds__(128) k_contract_bwd(const float* __restrict__ S, const float* __restrict__ dd,
                                                      const float* __restrict__ Wc, float* __restrict__ dS, int natom) {
  __shared__ float sWc[3 * NWV * NORB];
  __shared__ float sS[NANGC * NWV];
  __shared__ float sdd[NORB];
  __shared__ float G[NANGC * NORB];
  int i = blockIdx.x, tid = threadIdx.x;
  for (int idx = tid; idx < 3 * NWV * NORB; idx += 128) sWc[idx] = Wc[idx];
  for (int idx = tid; idx < NANGC * NWV; idx += 128) sS[idx] = S[(size_t)i * SROW + idx];
  for (int idx = tid; idx < NORB; idx += 128) sdd[idx] = dd[(size_t)i * NORB + idx];
  __syncthreads();
  for (int idx = tid; idx < NANGC * NORB; idx += 128) {
    int m = idx / NORB, o = idx - m * NORB;
    int L = (m == 0) ? 0 : ((m < 4) ? 1 : 2);
    float h = 0.f;
#pragma unroll
    for (int k = 0; k < NWV; ++k) h += sS[m * NWV + k] * sWc[L * (NWV * NORB) + k * NORB + o];
    G[idx] = 2.f * h * sdd[o];
  }
  __syncthreads();
  for (int idx = tid; idx < NANGC * NWV; idx += 128) {
    int m = idx / NWV, k = idx - m * NWV;
    int L = (m == 0) ? 0 : ((m < 4) ? 1 : 2);
    float a = 0.f;
    for (int o = 0; o < NORB; ++o) a += G[m * NORB + o] * sWc[L * (NWV * NORB) + k * NORB + o];
    dS[(size_t)i * SROW + idx] = a;
  }
}

// ---------------- WMMA MLP (16 atoms/block, 8 wave32s, both species with masked writes) ----------------
struct MlpP {
  const float *W1, *b1, *g1, *hh1, *a1, *s1, *W2, *b2, *g2, *hh2, *a2, *s2, *Wo, *bo;
};

// MODE 0: oc forward  -> o0 = c1 (coeff + delta)
// MODE 1: nn fwd+bwd  -> o0 = ddensity1, o1 = per-atom E, o2 = &E_sum
// MODE 2: oc recompute-fwd + bwd with upstream dcin -> o0 = ddensity0
template <int MODE>
__global__ void __launch_bounds__(256) k_mlp(MlpP p, const float* __restrict__ xin,
                                             const int* __restrict__ species, const float* __restrict__ coeff,
                                             const float* __restrict__ dcin, float* __restrict__ o0,
                                             float* __restrict__ o1, float* __restrict__ o2, int natom) {
  __shared__ _Float16 xh[16 * XLD];
  __shared__ _Float16 hh[16 * HLD];
  __shared__ _Float16 W1t[NLH * W1TLD];                       // W1[t]^T : [n][k], k padded 84->96
  __shared__ _Float16 W2t[NLH * W2TLD];                       // W2[t]^T : [n][k]
  __shared__ _Float16 W1h[(MODE >= 1) ? 96 * WHLD : 1];       // W1[t] row-major (backward only)
  __shared__ _Float16 W2h[(MODE >= 1) ? NLH * WHLD : 1];      // W2[t] row-major (backward only)
  __shared__ float z1s[16 * ZLD];
  __shared__ float z2s[16 * ZLD];
  __shared__ float bA[16 * ZLD];
  __shared__ float bB[16 * ZLD];
  __shared__ float mu1[16], rst1[16], mu2[16], rst2[16], red1[16], red2[16];
  __shared__ int sp[16];
  __shared__ float dcs[16 * 8];

  const int tid = threadIdx.x, lane = tid & 31, wv = tid >> 5;
  const int base = blockIdx.x * 16;

  for (int idx = tid; idx < 16 * 96; idx += 256) {
    int m = idx / 96, k = idx - m * 96, a = base + m;
    float v = (k < NORB && a < natom) ? xin[(size_t)a * NORB + k] : 0.f;
    xh[m * XLD + k] = (_Float16)v;
  }
  if (tid < 16) sp[tid] = (base + tid < natom) ? species[base + tid] : -1;
  if (MODE == 2) {
    for (int idx = tid; idx < 16 * 7; idx += 256) {
      int m = idx / 7, q = idx - m * 7, a = base + m;
      dcs[m * 8 + q] = (a < natom) ? dcin[(size_t)a * 7 + q] : 0.f;
    }
  }

  for (int t = 0; t < 2; ++t) {
    // ---- stage this type's weights into LDS (f16; coalesced global reads, LDS-side scatter) ----
    {
      const float* W1g = p.W1 + (size_t)t * NORB * NLH;
      const float* W2g = p.W2 + (size_t)t * NLH * NLH;
      for (int idx = tid; idx < NORB * NLH; idx += 256) {
        int k = idx >> 7, n = idx & 127;
        W1t[n * W1TLD + k] = (_Float16)W1g[idx];
      }
      for (int idx = tid; idx < NLH * 12; idx += 256) {       // zero K pad 84..95
        int n = idx / 12, k = NORB + (idx - n * 12);
        W1t[n * W1TLD + k] = (_Float16)0.f;
      }
      for (int idx = tid; idx < NLH * NLH; idx += 256) {
        int k = idx >> 7, n = idx & 127;
        W2t[n * W2TLD + k] = (_Float16)W2g[idx];
      }
      if (MODE >= 1) {
        for (int idx = tid; idx < 96 * NLH; idx += 256) {
          int j = idx >> 7, n = idx & 127;
          W1h[j * WHLD + n] = (j < NORB) ? (_Float16)W1g[(size_t)j * NLH + n] : (_Float16)0.f;
        }
        for (int idx = tid; idx < NLH * NLH; idx += 256) {
          int j = idx >> 7, n = idx & 127;
          W2h[j * WHLD + n] = (_Float16)W2g[idx];
        }
      }
    }
    __syncthreads();

    // layer1: z1 = x @ W1[t] + b1[t]   (K padded 84->96, zeros in both operands)
    { v8f acc = gemm16<3>(xh, XLD, W1t, W1TLD, wv * 16, lane);
      stD(z1s, ZLD, wv * 16, lane, acc, p.b1 + t * NLH); }
    __syncthreads();
    if (tid < 16) {
      float s = 0.f, ss = 0.f;
      for (int n = 0; n < NLH; ++n) { float v = z1s[tid * ZLD + n]; s += v; ss += v * v; }
      float m_ = s * (1.f / NLH);
      mu1[tid] = m_; rst1[tid] = rsqrtf(ss * (1.f / NLH) - m_ * m_ + 1e-5f);
    }
    __syncthreads();
    for (int idx = tid; idx < 16 * NLH; idx += 256) {
      int m = idx >> 7, n = idx & 127;
      float xc = (z1s[m * ZLD + n] - mu1[m]) * rst1[m];
      float zt = xc * p.g1[t * NLH + n] + p.hh1[t * NLH + n];
      float uu = p.s1[t * NLH + n] * zt;
      hh[m * HLD + n] = (_Float16)(p.a1[t * NLH + n] * uu * sigm(uu));
    }
    __syncthreads();
    // layer2: z2 = h1 @ W2[t] + b2[t]
    { v8f acc = gemm16<4>(hh, HLD, W2t, W2TLD, wv * 16, lane);
      stD(z2s, ZLD, wv * 16, lane, acc, p.b2 + t * NLH); }
    __syncthreads();
    if (tid < 16) {
      float s = 0.f, ss = 0.f;
      for (int n = 0; n < NLH; ++n) { float v = z2s[tid * ZLD + n]; s += v; ss += v * v; }
      float m_ = s * (1.f / NLH);
      mu2[tid] = m_; rst2[tid] = rsqrtf(ss * (1.f / NLH) - m_ * m_ + 1e-5f);
    }
    __syncthreads();
    for (int idx = tid; idx < 16 * NLH; idx += 256) {
      int m = idx >> 7, n = idx & 127;
      float xc = (z2s[m * ZLD + n] - mu2[m]) * rst2[m];
      float zt = xc * p.g2[t * NLH + n] + p.hh2[t * NLH + n];
      float uu = p.s2[t * NLH + n] * zt;
      bA[m * ZLD + n] = p.a2[t * NLH + n] * uu * sigm(uu);  // h2 f32
    }
    __syncthreads();
    // output head (VALU, tiny)
    if (MODE == 0) {
      if (tid < 16 * 7) {
        int m = tid / 7, q = tid - m * 7;
        float s = p.bo[t * 7 + q];
        for (int n = 0; n < NLH; ++n) s += bA[m * ZLD + n] * p.Wo[(size_t)t * NLH * 7 + n * 7 + q];
        if (sp[m] == t) o0[(size_t)(base + m) * NWV + q] = coeff[t * NWV + q] + s;
      }
    } else if (MODE == 1) {
      if (tid < 16) {
        int m = tid;
        float s = p.bo[t];
        for (int n = 0; n < NLH; ++n) s += bA[m * ZLD + n] * p.Wo[(size_t)t * NLH + n];
        if (sp[m] == t) { o1[base + m] = s; unsafeAtomicAdd(o2, s); }
      }
    }
    __syncthreads();

    if (MODE >= 1) {
      // upstream dh2 -> t1 (layer2 elementwise bwd) in bA
      for (int idx = tid; idx < 16 * NLH; idx += 256) {
        int m = idx >> 7, n = idx & 127;
        float dh;
        if (MODE == 1) dh = p.Wo[(size_t)t * NLH + n];  // dE/dout = 1
        else {
          dh = 0.f;
#pragma unroll
          for (int q = 0; q < 7; ++q) dh += dcs[m * 8 + q] * p.Wo[(size_t)t * NLH * 7 + n * 7 + q];
        }
        float xc = (z2s[m * ZLD + n] - mu2[m]) * rst2[m];
        float zt = xc * p.g2[t * NLH + n] + p.hh2[t * NLH + n];
        float uu = p.s2[t * NLH + n] * zt, sg = sigm(uu);
        float dsl = sg * (1.f + uu * (1.f - sg));
        bA[m * ZLD + n] = dh * p.a2[t * NLH + n] * dsl * p.s2[t * NLH + n] * p.g2[t * NLH + n];
      }
      __syncthreads();
      if (tid < 16) {
        float s1v = 0.f, s2v = 0.f;
        for (int n = 0; n < NLH; ++n) {
          float t1 = bA[tid * ZLD + n];
          float xc = (z2s[tid * ZLD + n] - mu2[tid]) * rst2[tid];
          s1v += t1; s2v += t1 * xc;
        }
        red1[tid] = s1v * (1.f / NLH); red2[tid] = s2v * (1.f / NLH);
      }
      __syncthreads();
      for (int idx = tid; idx < 16 * NLH; idx += 256) {
        int m = idx >> 7, n = idx & 127;
        float xc = (z2s[m * ZLD + n] - mu2[m]) * rst2[m];
        hh[m * HLD + n] = (_Float16)(rst2[m] * (bA[m * ZLD + n] - red1[m] - xc * red2[m]));  // dz2 f16
      }
      __syncthreads();
      // dh1 = dz2 @ W2^T  (B[k=n][j] = W2[j][n], contiguous rows of W2h)
      { v8f acc = gemm16<4>(hh, HLD, W2h, WHLD, wv * 16, lane);
        stD(bB, ZLD, wv * 16, lane, acc, nullptr); }
      __syncthreads();
      // layer1 elementwise bwd
      for (int idx = tid; idx < 16 * NLH; idx += 256) {
        int m = idx >> 7, n = idx & 127;
        float xc = (z1s[m * ZLD + n] - mu1[m]) * rst1[m];
        float zt = xc * p.g1[t * NLH + n] + p.hh1[t * NLH + n];
        float uu = p.s1[t * NLH + n] * zt, sg = sigm(uu);
        float dsl = sg * (1.f + uu * (1.f - sg));
        bB[m * ZLD + n] = bB[m * ZLD + n] * p.a1[t * NLH + n] * dsl * p.s1[t * NLH + n] * p.g1[t * NLH + n];
      }
      __syncthreads();
      if (tid < 16) {
        float s1v = 0.f, s2v = 0.f;
        for (int n = 0; n < NLH; ++n) {
          float t1 = bB[tid * ZLD + n];
          float xc = (z1s[tid * ZLD + n] - mu1[tid]) * rst1[tid];
          s1v += t1; s2v += t1 * xc;
        }
        red1[tid] = s1v * (1.f / NLH); red2[tid] = s2v * (1.f / NLH);
      }
      __syncthreads();
      for (int idx = tid; idx < 16 * NLH; idx += 256) {
        int m = idx >> 7, n = idx & 127;
        float xc = (z1s[m * ZLD + n] - mu1[m]) * rst1[m];
        hh[m * HLD + n] = (_Float16)(rst1[m] * (bB[m * ZLD + n] - red1[m] - xc * red2[m]));  // dz1 f16
      }
      __syncthreads();
      // dx = dz1 @ W1^T  (rows of W1h; cols 84..95 discarded) — waves 0..5
      if (wv < 6) {
        v8f acc = gemm16<4>(hh, HLD, W1h, WHLD, wv * 16, lane);
        int n = lane & 15, col = wv * 16 + n, mb = ((lane >> 4) & 1) * 8;
        if (col < NORB) {
#pragma unroll
          for (int r = 0; r < 8; ++r) {
            int m = mb + r;
            if (sp[m] == t) o0[(size_t)(base + m) * NORB + col] = acc[r];
          }
        }
      }
      __syncthreads();
    } else {
      __syncthreads();  // protect W1t/W2t before restaging for next t
    }
  }
}

// ---------------- host launch ----------------
static MlpP mkp(void* const* d, int b) {
  MlpP p;
  p.W1 = (const float*)d[b + 0];  p.b1 = (const float*)d[b + 1];
  p.g1 = (const float*)d[b + 2];  p.hh1 = (const float*)d[b + 3];
  p.a1 = (const float*)d[b + 4];  p.s1 = (const float*)d[b + 5];
  p.W2 = (const float*)d[b + 6];  p.b2 = (const float*)d[b + 7];
  p.g2 = (const float*)d[b + 8];  p.hh2 = (const float*)d[b + 9];
  p.a2 = (const float*)d[b + 10]; p.s2 = (const float*)d[b + 11];
  p.Wo = (const float*)d[b + 12]; p.bo = (const float*)d[b + 13];
  return p;
}

extern "C" void kernel_launch(void* const* d_in, const int* in_sizes, int n_in,
                              void* d_out, int out_size, void* d_ws, size_t ws_size,
                              hipStream_t stream) {
  const float* cart  = (const float*)d_in[0];
  const int*   ai    = (const int*)d_in[1];   // [0..P) = center, [P..2P) = neighbor
  const int*   ls    = (const int*)d_in[2];
  const int*   ns    = (const int*)d_in[3];
  const float* rs    = (const float*)d_in[4];
  const float* inta  = (const float*)d_in[5];
  const float* coeff = (const float*)d_in[6];
  const float* Wc    = (const float*)d_in[7];
  MlpP oc = mkp(d_in, 8);
  MlpP nn = mkp(d_in, 22);

  const int N = in_sizes[0] / 3;
  const int P = in_sizes[3];

  float* w = (float*)d_ws;
  size_t nS = (size_t)N * SROW, nD = (size_t)N * NORB, nC = (((size_t)N * NWV + 3) & ~(size_t)3);
  float* S0  = w;          float* S1  = S0 + nS;
  float* den0 = S1 + nS;   float* den1 = den0 + nD;
  float* c1  = den1 + nD;  float* dd1 = c1 + nC;
  float* dd0 = dd1 + nD;   float* dS1 = dd0 + nD;
  float* dS0 = dS1 + nS;   float* dc1 = dS0 + nS;
  float* c0a = dc1 + nC;

  float* outp = (float*)d_out;
  float* F    = outp + 1;
  float* perE = outp + 1 + (size_t)3 * N;

  int zgrid = (int)((2 * nS + 255) / 256);
  int pg = (P + 255) / 256;
  int mg = (N + 15) / 16;

  k_init<<<zgrid, 256, 0, stream>>>(S0, dc1, outp, c0a, coeff, ls, N);
  // forward
  k_pair_accum<<<pg, 256, 0, stream>>>(cart, ai, ns, rs, inta, c0a, S0, P);
  k_contract<<<N, 128, 0, stream>>>(S0, Wc, den0, N);
  k_mlp<0><<<mg, 256, 0, stream>>>(oc, den0, ls, coeff, nullptr, c1, nullptr, nullptr, N);
  k_pair_accum<<<pg, 256, 0, stream>>>(cart, ai, ns, rs, inta, c1, S1, P);
  k_contract<<<N, 128, 0, stream>>>(S1, Wc, den1, N);
  // nn head: energy + per-atom energies + ddensity1
  k_mlp<1><<<mg, 256, 0, stream>>>(nn, den1, ls, nullptr, nullptr, dd1, perE, outp, N);
  // backward
  k_contract_bwd<<<N, 128, 0, stream>>>(S1, dd1, Wc, dS1, N);
  k_pair_bwdA<<<pg, 256, 0, stream>>>(cart, ai, ns, rs, inta, dS1, dc1, P);
  k_mlp<2><<<mg, 256, 0, stream>>>(oc, den0, ls, nullptr, dc1, dd0, nullptr, nullptr, N);
  k_contract_bwd<<<N, 128, 0, stream>>>(S0, dd0, Wc, dS0, N);
  k_pair_bwdB<<<pg, 256, 0, stream>>>(cart, ai, ns, rs, inta, dS1, dS0, c1, c0a, F, P);
}
